// MultiHeadAttention_45638322487939
// MI455X (gfx1250) — matrix operand order repaired and checked by reading, not compiled, not run on hardware
//
#include <hip/hip_runtime.h>
#include <hip/hip_bf16.h>
#include <stdint.h>

// ---------------- problem constants ----------------
#define D_MODEL  1024
#define N_HEADS  16
#define HEAD_DIM 64
#define BATCH    4
#define SEQ      2048
#define M1       (BATCH * SEQ)        // 8192 rows of x / y
#define KT1      (D_MODEL / 32)       // 32 k-tiles (K = 1024, 32 per WMMA)
#define NT_QKV   (3 * D_MODEL / 16)   // 192 n-tiles for qkv projection
#define NT_OUT   (D_MODEL / 16)       // 64 n-tiles for out projection
#define MTILES   (M1 / 16)            // 512 m-tiles

typedef __attribute__((ext_vector_type(16))) _Float16 v16h;
typedef __attribute__((ext_vector_type(8)))  float    v8f;

union V16H { v16h v; uint32_t u[8]; _Float16 h[16]; };

__device__ __forceinline__ v8f zero_v8f() {
  v8f z = {0.f, 0.f, 0.f, 0.f, 0.f, 0.f, 0.f, 0.f};
  return z;
}

// A-matrix (16x32 f16, MxK) per-lane element->K mapping used here (contiguous k order, not the ISA table order):
// lane group g = lane/16, element e in v16h:
//   K = e + 16*g
__device__ __forceinline__ int ka_idx(int e, int g) {
  return e + 16 * g;
}
// B-matrix (32x16 f16, KxN): lane n = lane%16, element e -> K = e + 16*(lane/16)

// Issue one thread's share (2 x 16B) of an 8KB B-tile async copy into LDS.
// Each wave issues exactly 2 GLOBAL_LOAD_ASYNC_TO_LDS_B128 -> ASYNCcnt += 2.
__device__ __forceinline__ void issue_btile_async(const _Float16* __restrict__ src,
                                                  _Float16* dst_lds, int tid) {
#pragma unroll
  for (int i = 0; i < 2; ++i) {
    const _Float16* g = src + i * 2048 + tid * 8;   // 16B per lane, contiguous
    _Float16* l = dst_lds + i * 2048 + tid * 8;
    uint32_t lds_addr = (uint32_t)(uintptr_t)l;     // low 32 bits of flat = LDS offset
    unsigned long long gaddr = (unsigned long long)(uintptr_t)g;
    asm volatile("global_load_async_to_lds_b128 %0, %1, off"
                 :: "v"(lds_addr), "v"(gaddr) : "memory");
  }
}

// ---------------- packing kernels ----------------
// Pack fp32 row-major [M1, D_MODEL] into A-fragment order:
// dst[(mt*KT1 + kt)*512 + lane*16 + e] = (f16) src[mt*16 + lane%16][kt*32 + ka_idx(e, lane/16)]
__global__ void pack_a_f32_kernel(const float* __restrict__ src, _Float16* __restrict__ dst) {
  uint32_t idx  = blockIdx.x * 256u + threadIdx.x;   // < MTILES*KT1*512
  uint32_t tile = idx >> 9;
  uint32_t r    = idx & 511u;
  int lane = r >> 4, e = r & 15;
  int mt = tile / KT1, kt = tile - mt * KT1;
  int m = mt * 16 + (lane & 15);
  int k = kt * 32 + ka_idx(e, lane >> 4);
  dst[idx] = (_Float16)src[(size_t)m * D_MODEL + k];
}

// Same but f16 source (y re-pack for the out projection)
__global__ void pack_a_f16_kernel(const _Float16* __restrict__ src, _Float16* __restrict__ dst) {
  uint32_t idx  = blockIdx.x * 256u + threadIdx.x;
  uint32_t tile = idx >> 9;
  uint32_t r    = idx & 511u;
  int lane = r >> 4, e = r & 15;
  int mt = tile / KT1, kt = tile - mt * KT1;
  int m = mt * 16 + (lane & 15);
  int k = kt * 32 + ka_idx(e, lane >> 4);
  dst[idx] = src[(size_t)m * D_MODEL + k];
}

// Pack torch-style weight W [Nout, K] fp32 row-major into B-fragment order of W^T:
// dst[(kt*Ntiles + nt)*512 + lane*16 + e] = (f16) W[nt*16 + lane%16][kt*32 + e + 16*(lane/16)]
__global__ void pack_b_w_kernel(const float* __restrict__ w, _Float16* __restrict__ dst,
                                int Ntiles) {
  uint32_t idx  = blockIdx.x * 256u + threadIdx.x;
  uint32_t tile = idx >> 9;
  uint32_t r    = idx & 511u;
  int lane = r >> 4, e = r & 15;
  int kt = tile / Ntiles, nt = tile - kt * Ntiles;
  int n = nt * 16 + (lane & 15);
  int k = kt * 32 + e + 16 * (lane >> 4);
  dst[idx] = (_Float16)w[(size_t)n * D_MODEL + k];
}

// ---------------- QKV projection GEMM ----------------
// C[8192,3072] = x @ qkv_w^T + b. B tiles (shared by all 8 waves of the block)
// are staged in LDS by the async-DMA path, double buffered; A fragments are
// per-wave and loaded directly from global.
__global__ __launch_bounds__(256) void qkv_gemm_kernel(
    const _Float16* __restrict__ xA, const _Float16* __restrict__ wB,
    const float* __restrict__ bias,
    _Float16* __restrict__ Qrow, _Float16* __restrict__ Krow, _Float16* __restrict__ Vt) {
  __shared__ __align__(32) _Float16 sB[2][8 * 512];   // 2 x 8KB ping-pong
  int tid = threadIdx.x;
  int lane = tid & 31, wid = tid >> 5;
  int mt  = blockIdx.x * 8 + wid;     // 0..511
  int ntg = blockIdx.y;               // 0..23, 8 n-tiles each

  const _Float16* Ap = xA + ((size_t)mt * KT1) * 512 + lane * 16;
  const _Float16* Bbase = wB + (size_t)ntg * 8 * 512;

  v8f acc[8];
#pragma unroll
  for (int j = 0; j < 8; ++j) acc[j] = zero_v8f();

  issue_btile_async(Bbase, sB[0], tid);               // tile 0 -> buf 0

  for (int kt = 0; kt < KT1; ++kt) {
    int cur = kt & 1;
    if (kt + 1 < KT1) {
      issue_btile_async(Bbase + (size_t)(kt + 1) * NT_QKV * 512, sB[cur ^ 1], tid);
      asm volatile("s_wait_asynccnt 0x2" ::: "memory");   // tile kt landed (ours)
    } else {
      asm volatile("s_wait_asynccnt 0x0" ::: "memory");
    }
    __syncthreads();                                      // all waves' DMA landed

    v16h a = *reinterpret_cast<const v16h*>(Ap + (size_t)kt * 512);
    __builtin_prefetch(Ap + (size_t)(kt + 1) * 512, 0, 3);

    v16h bfr[8];
#pragma unroll
    for (int j = 0; j < 8; ++j)
      bfr[j] = *reinterpret_cast<const v16h*>(&sB[cur][j * 512 + lane * 16]);
#pragma unroll
    for (int j = 0; j < 8; ++j)
      acc[j] = __builtin_amdgcn_wmma_f32_16x16x32_f16(false, a, false, bfr[j],
                                                      (short)0, acc[j], false, false);
    __syncthreads();                                      // done reading buf before reuse
  }

  // epilogue: C layout -> lane holds n = lane%16, rows m = r + 8*(lane/16)
  int g = lane >> 4;
#pragma unroll
  for (int j = 0; j < 8; ++j) {
    int n = (ntg * 8 + j) * 16 + (lane & 15);    // 0..3071
    float bv = bias[n];
    int chunk = n >> 10;                          // 0=q 1=k 2=v
    int col = n & (D_MODEL - 1);
    int h = col >> 6, d = col & 63;
#pragma unroll
    for (int r = 0; r < 8; ++r) {
      int m = mt * 16 + r + 8 * g;
      int t = m & (SEQ - 1);
      int bb = m >> 11;
      float v = acc[j][r] + bv;
      size_t head = (size_t)(bb * N_HEADS + h);
      if (chunk == 0)      Qrow[(head * SEQ + t) * 64 + d] = (_Float16)(v * 0.125f); // fold 1/sqrt(64)
      else if (chunk == 1) Krow[(head * SEQ + t) * 64 + d] = (_Float16)v;
      else                 Vt[(head * 64 + d) * SEQ + t]   = (_Float16)v;
    }
  }
}

// ---------------- flash attention ----------------
// One wave: 16 query rows, full head (Dh=64). Online softmax, 32 keys / iter.
__global__ __launch_bounds__(256) void attn_kernel(
    const _Float16* __restrict__ Qrow, const _Float16* __restrict__ Krow,
    const _Float16* __restrict__ Vt, _Float16* __restrict__ y) {
  __shared__ _Float16 sP[8][16 * 32];   // per-wave P staging (C-layout -> A-fragment)

  int lane = threadIdx.x & 31, wid = threadIdx.x >> 5;
  int wg = blockIdx.x * 8 + wid;        // 0..8191
  int qt = wg & 127;                    // query tile (T/16 = 128)
  int h  = (wg >> 7) & (N_HEADS - 1);
  int b  = wg >> 11;
  int ln = lane & 15, g = lane >> 4;

  const _Float16* Qb = Qrow + ((size_t)(b * N_HEADS + h) * SEQ) * 64;
  const _Float16* Kb = Krow + ((size_t)(b * N_HEADS + h) * SEQ) * 64;
  const _Float16* Vb = Vt   + ((size_t)(b * N_HEADS + h) * 64) * SEQ;

  // Q A-fragments (Dh=64 -> 2 fragments of K=32), loaded once
  V16H qa[2];
  int mrow = qt * 16 + ln;
#pragma unroll
  for (int kt = 0; kt < 2; ++kt)
#pragma unroll
    for (int p = 0; p < 8; ++p) {
      int k = kt * 32 + ka_idx(2 * p, g);    // pairs (k, k+1) contiguous
      qa[kt].u[p] = *reinterpret_cast<const uint32_t*>(Qb + (size_t)mrow * 64 + k);
    }

  v8f o[4];
#pragma unroll
  for (int nt = 0; nt < 4; ++nt) o[nt] = zero_v8f();
  float mrun[8], lrun[8];
#pragma unroll
  for (int r = 0; r < 8; ++r) { mrun[r] = -1e30f; lrun[r] = 0.f; }

  int j2max = (qt * 16 + 15) >> 5;   // causal: keys <= qt*16+15
  for (int j2 = 0; j2 <= j2max; ++j2) {
    // ---- S = Q K^T for 32 keys (two 16-key sub-blocks) ----
    v8f s0 = zero_v8f(), s1 = zero_v8f();
#pragma unroll
    for (int kt = 0; kt < 2; ++kt) {
      v16h kb0 = *reinterpret_cast<const v16h*>(Kb + (size_t)(j2 * 32 + ln) * 64 + kt * 32 + 16 * g);
      v16h kb1 = *reinterpret_cast<const v16h*>(Kb + (size_t)(j2 * 32 + 16 + ln) * 64 + kt * 32 + 16 * g);
      s0 = __builtin_amdgcn_wmma_f32_16x16x32_f16(false, qa[kt].v, false, kb0,
                                                  (short)0, s0, false, false);
      s1 = __builtin_amdgcn_wmma_f32_16x16x32_f16(false, qa[kt].v, false, kb1,
                                                  (short)0, s1, false, false);
    }

    // ---- causal mask + online softmax (rows spread over 16 lanes) ----
    int key0 = j2 * 32 + ln, key1 = key0 + 16;
#pragma unroll
    for (int r = 0; r < 8; ++r) {
      int q = qt * 16 + r + 8 * g;
      float a0 = (key0 <= q) ? s0[r] : -1e30f;
      float a1 = (key1 <= q) ? s1[r] : -1e30f;
      float mx = fmaxf(a0, a1);
      mx = fmaxf(mx, __shfl_xor(mx, 1, 32));
      mx = fmaxf(mx, __shfl_xor(mx, 2, 32));
      mx = fmaxf(mx, __shfl_xor(mx, 4, 32));
      mx = fmaxf(mx, __shfl_xor(mx, 8, 32));
      float mnew = fmaxf(mrun[r], mx);
      float corr = __expf(mrun[r] - mnew);
      float p0 = __expf(a0 - mnew);
      float p1 = __expf(a1 - mnew);
      float rs = p0 + p1;
      rs += __shfl_xor(rs, 1, 32);
      rs += __shfl_xor(rs, 2, 32);
      rs += __shfl_xor(rs, 4, 32);
      rs += __shfl_xor(rs, 8, 32);
      lrun[r] = lrun[r] * corr + rs;
      mrun[r] = mnew;
#pragma unroll
      for (int nt = 0; nt < 4; ++nt) o[nt][r] *= corr;
      int row = r + 8 * g;
      sP[wid][row * 32 + ln]      = (_Float16)p0;
      sP[wid][row * 32 + 16 + ln] = (_Float16)p1;
    }
    asm volatile("s_wait_dscnt 0x0" ::: "memory");   // P stores visible to own wave

    // ---- reload P as one A-fragment (16 rows x 32 keys) ----
    V16H pa;
#pragma unroll
    for (int p = 0; p < 8; ++p) {
      int k = ka_idx(2 * p, g);
      pa.u[p] = *reinterpret_cast<const uint32_t*>(&sP[wid][ln * 32 + k]);
    }

    // ---- O += P V  (V stored transposed: lane-contiguous B fragments) ----
#pragma unroll
    for (int nt = 0; nt < 4; ++nt) {
      v16h vb = *reinterpret_cast<const v16h*>(Vb + (size_t)(nt * 16 + ln) * SEQ + j2 * 32 + 16 * g);
      o[nt] = __builtin_amdgcn_wmma_f32_16x16x32_f16(false, pa.v, false, vb,
                                                     (short)0, o[nt], false, false);
    }
  }

  // ---- normalize + store y row-major f16 [B,T,C] ----
#pragma unroll
  for (int r = 0; r < 8; ++r) {
    float inv = 1.0f / lrun[r];
    int t = qt * 16 + r + 8 * g;
    size_t base = ((size_t)(b * SEQ + t)) * D_MODEL + h * 64;
#pragma unroll
    for (int nt = 0; nt < 4; ++nt)
      y[base + nt * 16 + ln] = (_Float16)(o[nt][r] * inv);
  }
}

// ---------------- output projection GEMM ----------------
__global__ __launch_bounds__(256) void out_gemm_kernel(
    const _Float16* __restrict__ yA, const _Float16* __restrict__ wB,
    const float* __restrict__ bias, float* __restrict__ out) {
  __shared__ __align__(32) _Float16 sB[2][8 * 512];
  int tid = threadIdx.x;
  int lane = tid & 31, wid = tid >> 5;
  int mt  = blockIdx.x * 8 + wid;
  int ntg = blockIdx.y;               // 0..7

  const _Float16* Ap = yA + ((size_t)mt * KT1) * 512 + lane * 16;
  const _Float16* Bbase = wB + (size_t)ntg * 8 * 512;

  v8f acc[8];
#pragma unroll
  for (int j = 0; j < 8; ++j) acc[j] = zero_v8f();

  issue_btile_async(Bbase, sB[0], tid);

  for (int kt = 0; kt < KT1; ++kt) {
    int cur = kt & 1;
    if (kt + 1 < KT1) {
      issue_btile_async(Bbase + (size_t)(kt + 1) * NT_OUT * 512, sB[cur ^ 1], tid);
      asm volatile("s_wait_asynccnt 0x2" ::: "memory");
    } else {
      asm volatile("s_wait_asynccnt 0x0" ::: "memory");
    }
    __syncthreads();

    v16h a = *reinterpret_cast<const v16h*>(Ap + (size_t)kt * 512);
    __builtin_prefetch(Ap + (size_t)(kt + 1) * 512, 0, 3);

    v16h bfr[8];
#pragma unroll
    for (int j = 0; j < 8; ++j)
      bfr[j] = *reinterpret_cast<const v16h*>(&sB[cur][j * 512 + lane * 16]);
#pragma unroll
    for (int j = 0; j < 8; ++j)
      acc[j] = __builtin_amdgcn_wmma_f32_16x16x32_f16(false, a, false, bfr[j],
                                                      (short)0, acc[j], false, false);
    __syncthreads();
  }

  int g = lane >> 4;
#pragma unroll
  for (int j = 0; j < 8; ++j) {
    int n = (ntg * 8 + j) * 16 + (lane & 15);
    float bv = bias[n];
#pragma unroll
    for (int r = 0; r < 8; ++r) {
      int m = mt * 16 + r + 8 * g;
      out[(size_t)m * D_MODEL + n] = acc[j][r] + bv;
    }
  }
}

// ---------------- launch ----------------
extern "C" void kernel_launch(void* const* d_in, const int* in_sizes, int n_in,
                              void* d_out, int out_size, void* d_ws, size_t ws_size,
                              hipStream_t stream) {
  const float* x     = (const float*)d_in[0];
  const float* qkv_w = (const float*)d_in[1];
  const float* qkv_b = (const float*)d_in[2];
  const float* out_w = (const float*)d_in[3];
  const float* out_b = (const float*)d_in[4];
  // d_in[5] = mask: causal, handled analytically in the attention kernel.

  char* ws = (char*)d_ws;
  const size_t MB = 1u << 20;
  _Float16* xA    = (_Float16*)(ws);             // 16 MB  A-frags of x
  _Float16* wqkvB = (_Float16*)(ws + 16 * MB);   //  6 MB  B-frags of qkv_w^T
  _Float16* woutB = (_Float16*)(ws + 22 * MB);   //  2 MB  B-frags of out_w^T
  _Float16* Qrow  = (_Float16*)(ws + 24 * MB);   // 16 MB  [B,H,T,64] (pre-scaled)
  _Float16* Krow  = (_Float16*)(ws + 40 * MB);   // 16 MB  [B,H,T,64]
  _Float16* Vt    = (_Float16*)(ws + 56 * MB);   // 16 MB  [B,H,64,T] (transposed)
  _Float16* yrow  = (_Float16*)(ws + 72 * MB);   // 16 MB  [B,T,C] f16
  _Float16* yA    = (_Float16*)(ws + 88 * MB);   // 16 MB  A-frags of y  (total 104 MB)

  pack_a_f32_kernel<<<32768, 256, 0, stream>>>(x, xA);
  pack_b_w_kernel  <<<12288, 256, 0, stream>>>(qkv_w, wqkvB, NT_QKV);
  pack_b_w_kernel  <<< 4096, 256, 0, stream>>>(out_w, woutB, NT_OUT);
  qkv_gemm_kernel  <<<dim3(64, 24), 256, 0, stream>>>(xA, wqkvB, qkv_b, Qrow, Krow, Vt);
  attn_kernel      <<<1024, 256, 0, stream>>>(Qrow, Krow, Vt, yrow);
  pack_a_f16_kernel<<<32768, 256, 0, stream>>>(yrow, yA);
  out_gemm_kernel  <<<dim3(64, 8), 256, 0, stream>>>(yA, woutB, out_b, (float*)d_out);
}
